// DeceiverLT_8641474199940
// MI455X (gfx1250) — matrix operand order repaired and checked by reading, not compiled, not run on hardware
//
#include <hip/hip_runtime.h>
#include <hip/hip_bf16.h>
#include <math.h>

// ---------------------------------------------------------------------------
// Deceiver (Perceiver-style) forward for gfx1250 / MI455X.
//
// Heavy math runs through one generic batched GEMM kernel using
// v_wmma_f32_16x16x32_bf16 (fp32 inputs converted to bf16 via packed
// v_cvt_pk_bf16_f32, fp32 accumulation), software-pipelined: next K-tile is
// prefetched from global into registers while the current tile's WMMAs
// execute. Staging has a block-uniform fast path for interior blocks.
//
// Input flattening assumption: setup_inputs() dict in insertion order,
// nested dicts/lists depth-first:
//   0:x 1:latents 2..7:first{g,b,w1,b1,w2,b2} 8..13:l2l{g,b,qw,kvw,ow,ob}
//   then 4 layers x 26:
//     cross{gq,bq,gc,bc,qw,kvw,ow,ob} cff{g,b,w1,b1,w2,b2}
//     lattn{g,b,qw,kvw,ow,ob} lff{g,b,w1,b1,w2,b2}
//   118:h 119:w
// ---------------------------------------------------------------------------

typedef __attribute__((ext_vector_type(16))) __bf16 v16bf;
typedef __attribute__((ext_vector_type(8)))  float  v8f;
typedef __attribute__((ext_vector_type(2)))  float  v2f;
typedef __attribute__((ext_vector_type(2)))  __bf16 v2bf;

#define LN_EPS 1e-5f

// Packed fp32x2 -> bf16x2 (RNE): <2 x float> fptrunc matches
// v_cvt_pk_bf16_f32 on gfx1250 -- both halves converted in one issue.
__device__ __forceinline__ unsigned int pack_bf16x2(float lo, float hi) {
  v2f f; f[0] = lo; f[1] = hi;
  v2bf h = __builtin_convertvector(f, v2bf);
  union { v2bf v; unsigned int u; } cv; cv.v = h;
  return cv.u;
}

// ---------------------------------------------------------------------------
// Generic batched GEMM:  C = alpha * (A @ B)  [+ bias(col)] [+ addend]
//   A: M x K row-major (leading dim lda), fp32
//   B: K x N row-major (ldb)  -- or, if transB: B[k][n] = Bp[n*ldb + k]
//   C: M x N row-major (ldc), fp32
// blockIdx.z = batch; batch -> (b = z/H, h = z%H); pointer offsets
// b*s?b + h*s?h applied to A/B/C (addend shares C strides).
// Block: 256 threads = 8 waves; block tile 128(M) x 64(N); K-step 32.
// Wave w computes rows [w*16, w*16+16) x all 64 cols = 4 WMMA accumulators.
// ---------------------------------------------------------------------------
#define LSA_STRIDE 40   // half-words; row*80B -> 16B aligned, conflict-free frag reads
#define LSB_STRIDE 72   // half-words; row*144B -> 16B aligned

__global__ void __launch_bounds__(256)
gemm_bf16_kernel(const float* __restrict__ A, const float* __restrict__ Bm,
                 const float* __restrict__ bias, const float* __restrict__ addend,
                 float* __restrict__ C,
                 int M, int N, int K, int lda, int ldb, int ldc,
                 int H, long sAb, long sAh, long sBb, long sBh, long sCb, long sCh,
                 int transB, float alpha)
{
  __shared__ __align__(16) unsigned short lsA[128 * LSA_STRIDE];
  __shared__ __align__(16) unsigned short lsB[32 * LSB_STRIDE];

  const int z  = blockIdx.z;
  const int bb = z / H;
  const int hh = z - bb * H;
  A  += (long)bb * sAb + (long)hh * sAh;
  Bm += (long)bb * sBb + (long)hh * sBh;
  C  += (long)bb * sCb + (long)hh * sCh;
  if (addend) addend += (long)bb * sCb + (long)hh * sCh;

  const int m0   = blockIdx.x * 128;
  const int n0   = blockIdx.y * 64;
  const int tid  = threadIdx.x;
  const int lane = tid & 31;
  const int w    = tid >> 5;

  // staging coordinates (fixed per thread)
  const int ar  = tid >> 1;          // A LDS row 0..127
  const int aks = (tid & 1) * 16;    // A k-offset within tile: 0/16
  const int agr = m0 + ar;           // A global row
  const int bk  = tid >> 3;          // B (k-major) LDS k-row 0..31
  const int bns = (tid & 7) * 8;     // B n-offset 0..56
  const int tpr = tid >> 3;          // B^T n-pair index 0..31 (covers n0+2*tpr, +1)
  const int tkg = (tid & 7) * 4;     // B^T k group offset 0..28

  // Block-uniform fast-path conditions: with lda%4==0 and 16B-aligned base,
  // every thread's float4 address is aligned (k0%32==0, aks%16==0, bns%8==0,
  // n0%64==0, tkg%4==0).
  const bool fastA = ((lda & 3) == 0) &&
                     ((((unsigned long long)(const void*)A) & 15ull) == 0) &&
                     (m0 + 128 <= M);
  const bool fastB = ((ldb & 3) == 0) &&
                     ((((unsigned long long)(const void*)Bm) & 15ull) == 0) &&
                     (n0 + 64 <= N);

  const int nk = (K + 31) >> 5;

  float ta[16];   // prefetched A values (16 consecutive k of one row)
  float tbv[8];   // prefetched B values

  auto loadA = [&](int k0) {
    if (fastA && (k0 + 32 <= K)) {
      const float4* p4 = (const float4*)(A + (long)agr * lda + k0 + aks);
      float4 q0 = p4[0], q1 = p4[1], q2 = p4[2], q3 = p4[3];
      ta[0]=q0.x; ta[1]=q0.y; ta[2]=q0.z; ta[3]=q0.w;
      ta[4]=q1.x; ta[5]=q1.y; ta[6]=q1.z; ta[7]=q1.w;
      ta[8]=q2.x; ta[9]=q2.y; ta[10]=q2.z; ta[11]=q2.w;
      ta[12]=q3.x; ta[13]=q3.y; ta[14]=q3.z; ta[15]=q3.w;
    } else {
      #pragma unroll
      for (int i = 0; i < 16; i++) {
        int k = k0 + aks + i;
        ta[i] = (agr < M && k < K) ? A[(long)agr * lda + k] : 0.0f;
      }
    }
  };
  auto loadB = [&](int k0) {
    if (!transB) {
      // 8 consecutive n of one k row
      const int k = k0 + bk;
      if (fastB && (k0 + 32 <= K)) {
        const float4* p4 = (const float4*)(Bm + (long)k * ldb + n0 + bns);
        float4 q0 = p4[0], q1 = p4[1];
        tbv[0]=q0.x; tbv[1]=q0.y; tbv[2]=q0.z; tbv[3]=q0.w;
        tbv[4]=q1.x; tbv[5]=q1.y; tbv[6]=q1.z; tbv[7]=q1.w;
      } else {
        #pragma unroll
        for (int i = 0; i < 8; i++) {
          int n = n0 + bns + i;
          tbv[i] = (k < K && n < N) ? Bm[(long)k * ldb + n] : 0.0f;
        }
      }
    } else {
      // 4 consecutive k of a pair of adjacent n columns
      const int n = n0 + tpr * 2;
      if (fastB && (k0 + 32 <= K)) {
        float4 qa = *(const float4*)(Bm + (long)n * ldb + k0 + tkg);
        float4 qc = *(const float4*)(Bm + (long)(n + 1) * ldb + k0 + tkg);
        tbv[0]=qa.x; tbv[1]=qa.y; tbv[2]=qa.z; tbv[3]=qa.w;
        tbv[4]=qc.x; tbv[5]=qc.y; tbv[6]=qc.z; tbv[7]=qc.w;
      } else {
        #pragma unroll
        for (int j = 0; j < 2; j++) {
          #pragma unroll
          for (int i = 0; i < 4; i++) {
            int k = k0 + tkg + i;
            tbv[j*4 + i] = ((n + j) < N && k < K) ? Bm[(long)(n + j) * ldb + k] : 0.0f;
          }
        }
      }
    }
  };
  auto storeLDS = [&]() {
    unsigned int pk[8];
    #pragma unroll
    for (int i = 0; i < 8; i++)
      pk[i] = pack_bf16x2(ta[2*i], ta[2*i+1]);
    uint4* dst = (uint4*)&lsA[ar * LSA_STRIDE + aks];
    dst[0] = make_uint4(pk[0], pk[1], pk[2], pk[3]);
    dst[1] = make_uint4(pk[4], pk[5], pk[6], pk[7]);
    if (!transB) {
      unsigned int pb[4];
      #pragma unroll
      for (int i = 0; i < 4; i++)
        pb[i] = pack_bf16x2(tbv[2*i], tbv[2*i+1]);
      *(uint4*)&lsB[bk * LSB_STRIDE + bns] = make_uint4(pb[0], pb[1], pb[2], pb[3]);
    } else {
      // lsB[k][n] and lsB[k][n+1] are adjacent half-words -> one b32 store each
      #pragma unroll
      for (int i = 0; i < 4; i++)
        *(unsigned int*)&lsB[(tkg + i) * LSB_STRIDE + tpr * 2] =
            pack_bf16x2(tbv[i], tbv[4 + i]);
    }
  };

  v8f acc[4] = {};

  // pipeline prologue: prefetch tile 0
  loadA(0);
  loadB(0);

  for (int kt = 0; kt < nk; kt++) {
    // convert + store current tile to LDS
    storeLDS();
    __syncthreads();

    // prefetch next tile while WMMAs below execute
    if (kt + 1 < nk) { loadA((kt + 1) * 32); loadB((kt + 1) * 32); }

    // ---- fragments + WMMA ----
    // A frag (ISA 7.12.2, 16-bit A 16x32): lane&15 = M row;
    // lanes 0-15: elems 0..7 -> K 0..7, elems 8..15 -> K 16..23
    // lanes 16-31: +8 on both K groups.
    union { v16bf v; uint4 q[2]; } fa;
    {
      const int frow = w * 16 + (lane & 15);
      const int kbo  = (lane >> 4) * 8;
      fa.q[0] = *(const uint4*)&lsA[frow * LSA_STRIDE + kbo];
      fa.q[1] = *(const uint4*)&lsA[frow * LSA_STRIDE + kbo + 16];
    }
    #pragma unroll
    for (int nt = 0; nt < 4; nt++) {
      // B frag: lane&15 = N col; elements follow A's K order (table order)
      union { v16bf v; uint4 q[2]; unsigned short s[16]; } fb;
      {
        const int bn  = nt * 16 + (lane & 15);
        const int kbb = (lane >> 4) * 8;
        #pragma unroll
        for (int i = 0; i < 8; i++) {
          fb.s[i]     = lsB[(kbb + i) * LSB_STRIDE + bn];
          fb.s[8 + i] = lsB[(kbb + 16 + i) * LSB_STRIDE + bn];
        }
      }
      acc[nt] = __builtin_amdgcn_wmma_f32_16x16x32_bf16(
          false, fa.v, false, fb.v, (short)0, acc[nt], false, false);
    }
    __syncthreads();
  }

  // ---- epilogue: C/D layout: VGPR r -> M=r (+8 for lanes>=16), lane&15 -> N
  const int rbase = m0 + w * 16 + ((lane >> 4) << 3);
  const int cbase = n0 + (lane & 15);
  #pragma unroll
  for (int nt = 0; nt < 4; nt++) {
    const int col = cbase + nt * 16;
    if (col >= N) continue;
    const float bv = bias ? bias[col] : 0.0f;
    #pragma unroll
    for (int r = 0; r < 8; r++) {
      const int row = rbase + r;
      if (row < M) {
        float v = acc[nt][r] * alpha + bv;
        if (addend) v += addend[(long)row * ldc + col];
        C[(long)row * ldc + col] = v;
      }
    }
  }
}

// ---------------------------------------------------------------------------
// LayerNorm over last dim (one 128-thread block per row).
// ---------------------------------------------------------------------------
__global__ void __launch_bounds__(128)
layernorm_kernel(const float* __restrict__ x, const float* __restrict__ g,
                 const float* __restrict__ b, float* __restrict__ y, int dim)
{
  __shared__ float red[128];
  const long row = blockIdx.x;
  const float* xr = x + row * dim;
  float* yr = y + row * dim;
  const int t = threadIdx.x;
  float s1 = 0.0f, s2 = 0.0f;
  for (int c = t; c < dim; c += 128) { float v = xr[c]; s1 += v; s2 += v * v; }
  red[t] = s1; __syncthreads();
  for (int o = 64; o > 0; o >>= 1) { if (t < o) red[t] += red[t + o]; __syncthreads(); }
  const float mu = red[0] / (float)dim;
  __syncthreads();
  red[t] = s2; __syncthreads();
  for (int o = 64; o > 0; o >>= 1) { if (t < o) red[t] += red[t + o]; __syncthreads(); }
  const float var = red[0] / (float)dim - mu * mu;
  const float rinv = rsqrtf(var + LN_EPS);
  for (int c = t; c < dim; c += 128)
    yr[c] = (xr[c] - mu) * rinv * g[c] + b[c];
}

// ---------------------------------------------------------------------------
// Row softmax over exactly 512 columns (both attentions have 512 keys).
// ---------------------------------------------------------------------------
__global__ void __launch_bounds__(128)
softmax512_kernel(float* __restrict__ x)
{
  __shared__ float red[128];
  const long row = blockIdx.x;
  float* p = x + row * 512;
  const int t = threadIdx.x;
  float v[4];
  float m = -3.4e38f;
  #pragma unroll
  for (int i = 0; i < 4; i++) { v[i] = p[t + 128 * i]; m = fmaxf(m, v[i]); }
  red[t] = m; __syncthreads();
  for (int o = 64; o > 0; o >>= 1) { if (t < o) red[t] = fmaxf(red[t], red[t + o]); __syncthreads(); }
  m = red[0]; __syncthreads();
  float s = 0.0f;
  #pragma unroll
  for (int i = 0; i < 4; i++) { v[i] = expf(v[i] - m); s += v[i]; }
  red[t] = s; __syncthreads();
  for (int o = 64; o > 0; o >>= 1) { if (t < o) red[t] += red[t + o]; __syncthreads(); }
  const float inv = 1.0f / red[0];
  #pragma unroll
  for (int i = 0; i < 4; i++) p[t + 128 * i] = v[i] * inv;
}

// ---------------------------------------------------------------------------
// GEGLU gate: t[r,c] = h[r,c] * gelu_exact(h[r, halfN + c])
// ---------------------------------------------------------------------------
__global__ void geglu_kernel(const float* __restrict__ h, float* __restrict__ t,
                             int halfN, long total)
{
  const long idx = (long)blockIdx.x * blockDim.x + threadIdx.x;
  if (idx >= total) return;
  const long r = idx / halfN;
  const int  c = (int)(idx - r * halfN);
  const float* hr = h + r * (2L * halfN);
  const float a  = hr[c];
  const float gt = hr[halfN + c];
  const float ge = 0.5f * gt * (1.0f + erff(gt * 0.70710678118654752440f));
  t[idx] = a * ge;
}

// ---------------------------------------------------------------------------
// Fourier positional encoding -> data buffer (4, 16384, 29); channels 0..2 = 0.
// ---------------------------------------------------------------------------
__global__ void posenc_kernel(float* __restrict__ data)
{
  const long idx = (long)blockIdx.x * blockDim.x + threadIdx.x;
  if (idx >= 4L * 16384 * 29) return;
  const int  c   = (int)(idx % 29);
  const long pix = (idx / 29) % 16384;
  float val = 0.0f;
  if (c >= 3) {
    const int cc = c - 3;
    const int d  = cc / 13;      // axis 0 = rows(i), 1 = cols(j)
    const int tt = cc - d * 13;  // 0..5 sin, 6..11 cos, 12 raw pos
    const int i = (int)(pix >> 7), j = (int)(pix & 127);
    const float pos = -1.0f + 2.0f * (float)(d == 0 ? i : j) / 127.0f;
    if (tt == 12) {
      val = pos;
    } else {
      const int s = (tt < 6) ? tt : tt - 6;
      // scales = logspace(1, log2(5), 6, base=2)
      const float e = 1.0f + (float)s * ((2.3219280948873623f - 1.0f) / 5.0f);
      const float pe = pos * exp2f(e) * 3.14159265358979323846f;
      val = (tt < 6) ? sinf(pe) : cosf(pe);
    }
  }
  data[idx] = val;
}

// ---------------------------------------------------------------------------
// lat[b,n,d] = attn_row[b,d] + x_row[b,d] + latents[n,d]
// (l2l attention on identical broadcast tokens reduces exactly to v@ow+ob)
// ---------------------------------------------------------------------------
__global__ void init_lat_kernel(const float* __restrict__ attn,
                                const float* __restrict__ xb,
                                const float* __restrict__ latents,
                                float* __restrict__ lat)
{
  const long idx = (long)blockIdx.x * blockDim.x + threadIdx.x;
  if (idx >= 4L * 512 * 512) return;
  const int  d = (int)(idx % 512);
  const long n = (idx / 512) % 512;
  const long b = idx / (512L * 512);
  lat[idx] = attn[b * 512 + d] + xb[b * 512 + d] + latents[n * 512 + d];
}

// ---------------------------------------------------------------------------
// out[b,i,j,c] = data[b, i*128+j, c], c < 3
// ---------------------------------------------------------------------------
__global__ void writeout_kernel(const float* __restrict__ data, float* __restrict__ out)
{
  const long idx = (long)blockIdx.x * blockDim.x + threadIdx.x;
  if (idx >= 4L * 128 * 128 * 3) return;
  const int  c   = (int)(idx % 3);
  const long pix = (idx / 3) % 16384;
  const long b   = idx / (3L * 16384);
  out[idx] = data[(b * 16384 + pix) * 29 + c];
}

// ---------------------------------------------------------------------------
extern "C" void kernel_launch(void* const* d_in, const int* in_sizes, int n_in,
                              void* d_out, int out_size, void* d_ws, size_t ws_size,
                              hipStream_t stream)
{
  (void)in_sizes; (void)n_in; (void)out_size; (void)ws_size;
  auto in = [&](int i) { return (const float*)d_in[i]; };

  const float* x_in    = in(0);
  const float* latents = in(1);

  // ---- workspace layout (floats); ~200 MB total ----
  const long LAT_SZ  = 4L * 512 * 512;    // 1,048,576
  const long DATA_SZ = 4L * 16384 * 29;   // 1,900,544
  const long XN_SZ   = 1900544;           // max(2048*512, 65536*29)
  const long CN_SZ   = 1048576;
  const long Q_SZ    = 4194304;           // max(2048*512, 65536*64)
  const long KV_SZ   = 2097152;           // max(2048*1024, 2048*128)
  const long SIM_SZ  = 33554432;          // max(32*512*512, 4*16384*512)
  const long O_SZ    = 4194304;
  float* lat  = (float*)d_ws;
  float* data = lat  + LAT_SZ;
  float* xn   = data + DATA_SZ;
  float* cn   = xn   + XN_SZ;
  float* qb   = cn   + CN_SZ;
  float* kvb  = qb   + Q_SZ;
  float* simb = kvb  + KV_SZ;
  float* atno = simb + SIM_SZ;
  float* lns  = atno + O_SZ;      // 4x512
  float* vs   = lns  + 2048;      // 4x512
  float* as_  = vs   + 2048;      // 4x512
  float* xb   = as_  + 2048;      // 4x512
  // FF hidden/gated buffers alias the (then-dead) attention score buffer.
  float* hb = simb;               // up to 65536*232 = 15,204,352
  float* tb = simb + 15204352;    // up to 65536*116 =  7,602,176

  auto gemm = [&](const float* A, const float* Bm, const float* bias,
                  const float* addend, float* C, int M, int N, int K,
                  int lda, int ldb, int ldc, int batches, int H,
                  long sAb, long sAh, long sBb, long sBh, long sCb, long sCh,
                  int transB, float alpha) {
    dim3 grid((M + 127) / 128, (N + 63) / 64, batches);
    gemm_bf16_kernel<<<grid, 256, 0, stream>>>(A, Bm, bias, addend, C, M, N, K,
        lda, ldb, ldc, H, sAb, sAh, sBb, sBh, sCb, sCh, transB, alpha);
  };
  auto gemm0 = [&](const float* A, const float* Bm, const float* bias,
                   const float* addend, float* C, int M, int N, int K,
                   float alpha = 1.0f) {
    gemm(A, Bm, bias, addend, C, M, N, K, K, N, N, 1, 1,
         0, 0, 0, 0, 0, 0, 0, alpha);
  };
  auto EL = [&](long tot) { return (unsigned)((tot + 255) / 256); };

  // ---- positional encoding / data init ----
  posenc_kernel<<<EL(4L*16384*29), 256, 0, stream>>>(data);

  // ---- first GEGLU-FF on x (4 x 512) ----
  layernorm_kernel<<<4, 128, 0, stream>>>(x_in, in(2), in(3), lns, 512);
  gemm0(lns, in(4), in(5), nullptr, hb, 4, 4096, 512);
  geglu_kernel<<<EL(4L*2048), 256, 0, stream>>>(hb, tb, 2048, 4L*2048);
  gemm0(tb, in(6), in(7), x_in, xb, 4, 512, 2048);

  // ---- l2l attention on identical broadcast tokens: exact shortcut o = v ----
  layernorm_kernel<<<4, 128, 0, stream>>>(xb, in(8), in(9), lns, 512);
  gemm(lns, in(11) + 512, nullptr, nullptr, vs, 4, 512, 512,
       512, 1024, 512, 1, 1, 0, 0, 0, 0, 0, 0, 0, 1.0f);   // v = ln @ kvw[:,512:]
  gemm0(vs, in(12), in(13), nullptr, as_, 4, 512, 512);      // o = v @ ow + ob
  init_lat_kernel<<<EL(LAT_SZ), 256, 0, stream>>>(as_, xb, latents, lat);

  // ---- layers ----
  for (int li = 0; li < 4; li++) {
    const int p = 14 + 26 * li;
    const float *gq=in(p+0), *bq=in(p+1), *gc=in(p+2), *bc=in(p+3);
    const float *c_qw=in(p+4), *c_kvw=in(p+5), *c_ow=in(p+6), *c_ob=in(p+7);
    const float *cf_g=in(p+8), *cf_b=in(p+9), *cf_w1=in(p+10), *cf_b1=in(p+11);
    const float *cf_w2=in(p+12), *cf_b2=in(p+13);
    const float *la_g=in(p+14), *la_b=in(p+15), *la_qw=in(p+16), *la_kvw=in(p+17);
    const float *la_ow=in(p+18), *la_ob=in(p+19);
    const float *lf_g=in(p+20), *lf_b=in(p+21), *lf_w1=in(p+22), *lf_b1=in(p+23);
    const float *lf_w2=in(p+24), *lf_b2=in(p+25);

    // ---- latent self-attention (B=4, H=8, N=512, dh=64) ----
    layernorm_kernel<<<2048, 128, 0, stream>>>(lat, la_g, la_b, xn, 512);
    gemm0(xn, la_qw, nullptr, nullptr, qb, 2048, 512, 512);
    gemm0(xn, la_kvw, nullptr, nullptr, kvb, 2048, 1024, 512);
    // sim(b,h) = Q (512x64) @ K^T (64x512) * 1/8
    gemm(qb, kvb, nullptr, nullptr, simb, 512, 512, 64,
         512, 1024, 512, 32, 8,
         512L*512, 64, 512L*1024, 64, 8L*512*512, 512L*512, 1, 0.125f);
    softmax512_kernel<<<32 * 512, 128, 0, stream>>>(simb);
    // o(b,h) = P (512x512) @ V (512x64)
    gemm(simb, kvb + 512, nullptr, nullptr, atno, 512, 64, 512,
         512, 1024, 512, 32, 8,
         8L*512*512, 512L*512, 512L*1024, 64, 512L*512, 64, 0, 1.0f);
    gemm0(atno, la_ow, la_ob, lat, lat, 2048, 512, 512);     // lat += o@ow+ob

    // ---- latent GEGLU-FF ----
    layernorm_kernel<<<2048, 128, 0, stream>>>(lat, lf_g, lf_b, xn, 512);
    gemm0(xn, lf_w1, lf_b1, nullptr, hb, 2048, 4096, 512);
    geglu_kernel<<<EL(2048L*2048), 256, 0, stream>>>(hb, tb, 2048, 2048L*2048);
    gemm0(tb, lf_w2, lf_b2, lat, lat, 2048, 512, 2048);

    // ---- cross attention (B=4, H=1, 16384 queries, 512 keys, dh=64) ----
    layernorm_kernel<<<65536, 128, 0, stream>>>(data, gq, bq, xn, 29);   // dn
    layernorm_kernel<<<2048, 128, 0, stream>>>(lat, gc, bc, cn, 512);    // cn
    gemm0(xn, c_qw, nullptr, nullptr, qb, 65536, 64, 29);
    gemm0(cn, c_kvw, nullptr, nullptr, kvb, 2048, 128, 512);
    gemm(qb, kvb, nullptr, nullptr, simb, 16384, 512, 64,
         64, 128, 512, 4, 1,
         16384L*64, 0, 512L*128, 0, 16384L*512, 0, 1, 0.125f);
    softmax512_kernel<<<65536, 128, 0, stream>>>(simb);
    gemm(simb, kvb + 64, nullptr, nullptr, atno, 16384, 64, 512,
         512, 128, 64, 4, 1,
         16384L*512, 0, 512L*128, 0, 16384L*64, 0, 0, 1.0f);
    gemm0(atno, c_ow, c_ob, data, data, 65536, 29, 64);      // data += o@ow+ob

    // ---- cross GEGLU-FF (29 -> 232 -> 29) ----
    layernorm_kernel<<<65536, 128, 0, stream>>>(data, cf_g, cf_b, xn, 29);
    gemm0(xn, cf_w1, cf_b1, nullptr, hb, 65536, 232, 29);
    geglu_kernel<<<EL(65536L*116), 256, 0, stream>>>(hb, tb, 116, 65536L*116);
    gemm0(tb, cf_w2, cf_b2, data, data, 65536, 29, 116);
  }

  // ---- slice out first 3 channels ----
  writeout_kernel<<<EL(4L*128*128*3), 256, 0, stream>>>(data, (float*)d_out);
}
